// MultiHeadAttention_21698174780245
// MI455X (gfx1250) — compile-verified, matrix operand order checked by reading, not run on hardware
//
#include <hip/hip_runtime.h>

// ---------------------------------------------------------------------------
// MultiHeadAttention forward for MI455X (gfx1250, wave32, WMMA + TDM).
// All GEMMs use v_wmma_f32_16x16x32_f16; A-operands staged to LDS by the
// Tensor Data Mover (tensor_load_to_lds, TENSORcnt), double buffered.
// ---------------------------------------------------------------------------

#define D_MODEL 1024
#define NHEAD   16
#define DHEAD   64
#define BATCH   2
#define SEQ     2048
#define MROWS   (BATCH * SEQ)   // 4096 flattened (b,s) rows

typedef __attribute__((ext_vector_type(16))) _Float16 v16h;
typedef __attribute__((ext_vector_type(8)))  _Float16 v8h;
typedef __attribute__((ext_vector_type(8)))  float    v8f;
typedef __attribute__((ext_vector_type(4)))  unsigned u32x4;
typedef __attribute__((ext_vector_type(8)))  int      i32x8;
typedef __attribute__((ext_vector_type(4)))  int      i32x4;

#if defined(__HIP_DEVICE_COMPILE__) && __has_builtin(__builtin_amdgcn_tensor_load_to_lds)
#define USE_TDM 1
#else
#define USE_TDM 0
#endif

// --- WMMA operand packing ---------------------------------------------------
// 16-bit A (16x32) / B (32x16) lane layout (ISA 7.12.2):
//   lane L, half = L>>4, idx = L&15:
//   vector elems 0..7  <-> k = kBase + half*8 + i
//   vector elems 8..15 <-> k = kBase + 16 + half*8 + i
// So per lane we need two contiguous 8-element chunks at +0 and +16.

__device__ __forceinline__ v16h pack16_from_f32(const float* __restrict__ p) {
  v16h r;
#pragma unroll
  for (int i = 0; i < 8; ++i) {
    r[i]     = (_Float16)p[i];
    r[8 + i] = (_Float16)p[16 + i];
  }
  return r;
}

__device__ __forceinline__ v16h pack16_from_f16(const _Float16* __restrict__ p) {
  v8h c0 = *(const v8h*)(p);
  v8h c1 = *(const v8h*)(p + 16);
  v16h r;
#pragma unroll
  for (int i = 0; i < 8; ++i) { r[i] = c0[i]; r[8 + i] = c1[i]; }
  return r;
}

__device__ __forceinline__ v8f wmma_acc(v16h a, v16h b, v8f c) {
  // D = A*B + C ; emits v_wmma_f32_16x16x32_f16
  return __builtin_amdgcn_wmma_f32_16x16x32_f16(
      /*neg_a=*/false, a, /*neg_b=*/false, b,
      /*c_mod=*/(short)0, c, /*reuse_a=*/false, /*reuse_b=*/false);
}

__device__ __forceinline__ void wait_tensorcnt0() {
#if __has_builtin(__builtin_amdgcn_s_wait_tensorcnt)
  __builtin_amdgcn_s_wait_tensorcnt(0);
#else
  asm volatile("s_wait_tensorcnt 0x0" ::: "memory");
#endif
}

#if USE_TDM
// 2D tile load Global -> LDS via the Tensor Data Mover.
// D# group0: [1:0]=count=1, [63:32]=lds_addr, [120:64]=global_addr, [127:126]=type(2)
// D# group1: [17:16]=data_size(log2 bytes), [20]=pad_enable, [24:22]=pad_interval,
//            [31:25]=pad_amount, [79:48]=tensor_dim0, [111:80]=tensor_dim1,
//            [127:112]=tile_dim0, [143:128]=tile_dim1, [207:160]=tensor_dim0_stride
__device__ __forceinline__ void tdm_load_2d(unsigned ldsAddr, const void* gsrc,
                                            unsigned elemLog2, unsigned tileCols,
                                            unsigned tileRows, unsigned tensorLd,
                                            unsigned tensorRows,
                                            unsigned padIntervalCode,
                                            unsigned padAmountCode,
                                            unsigned padEnable) {
  unsigned long long ga = (unsigned long long)(uintptr_t)gsrc;
  u32x4 g0;
  g0[0] = 1u;                                  // count = 1 valid user descriptor
  g0[1] = ldsAddr;                             // LDS byte address of tile start
  g0[2] = (unsigned)ga;                        // global_addr[31:0]
  g0[3] = (unsigned)((ga >> 32) & 0x01FFFFFFull) | 0x80000000u; // [56:32] | type=2
  i32x8 g1;
  g1[0] = (int)((elemLog2 << 16) | (padEnable << 20) |
                (padIntervalCode << 22) | (padAmountCode << 25));
  g1[1] = (int)(tensorLd << 16);               // tensor_dim0[15:0] in bits 63:48
  g1[2] = (int)((tensorLd >> 16) | (tensorRows << 16));
  g1[3] = (int)((tensorRows >> 16) | (tileCols << 16));
  g1[4] = (int)(tileRows & 0xFFFFu);           // tile_dim1 ; tile_dim2 = 0
  g1[5] = (int)tensorLd;                       // tensor_dim0_stride[31:0]
  g1[6] = 0;
  g1[7] = 0;
  i32x4 z4;
  z4[0] = 0; z4[1] = 0; z4[2] = 0; z4[3] = 0;  // groups 2/3 unused (<=2D tensor)
  i32x8 z8;
#pragma unroll
  for (int i = 0; i < 8; ++i) z8[i] = 0;
  // 6-arg (clang-23 / therock) form: g0, g1, g2, g3, extra, cpol
  __builtin_amdgcn_tensor_load_to_lds(g0, g1, z4, z4, z8, 0);
}
#endif

// ---------------------------------------------------------------------------
// Kernel 0: W [K=1024, N=1024] fp32 -> WT [N, K] f16 (transpose + downconvert)
// so B-operand loads in every GEMM are contiguous per lane.
// ---------------------------------------------------------------------------
__global__ __launch_bounds__(256) void wt_f16_kernel(const float* __restrict__ W,
                                                     _Float16* __restrict__ WT) {
  __shared__ float tile[32][33];
  const int tx = threadIdx.x & 31;
  const int ty = threadIdx.x >> 5;          // 0..7
  const int bi = blockIdx.x & 31;           // row-tile of W (K)
  const int bj = blockIdx.x >> 5;           // col-tile of W (N)
  const int r0 = bi * 32, c0 = bj * 32;
#pragma unroll
  for (int k = 0; k < 4; ++k)
    tile[ty + 8 * k][tx] = W[(size_t)(r0 + ty + 8 * k) * D_MODEL + c0 + tx];
  __syncthreads();
#pragma unroll
  for (int k = 0; k < 4; ++k)
    WT[(size_t)(c0 + ty + 8 * k) * D_MODEL + r0 + tx] = (_Float16)tile[tx][ty + 8 * k];
}

// ---------------------------------------------------------------------------
// Kernel 1: projection  dst = X[4096,1024] @ W + b   (W given as WT f16 [N,K])
// All 8 waves of a block share one 16-row A stripe -> TDM-stage A tiles
// (16x32 fp32, +1 dword row pad => LDS stride 33) with double buffering.
// transposedV == 0 : dst f16 [B, H, S, DHEAD]   (q, k)
// transposedV == 1 : dst f16 [B, H, DHEAD, S]   (v, pre-transposed for attn@V)
// ---------------------------------------------------------------------------
__global__ __launch_bounds__(256) void proj_kernel(const float* __restrict__ X,
                                                   const _Float16* __restrict__ WT,
                                                   const float* __restrict__ bias,
                                                   _Float16* __restrict__ dst,
                                                   int transposedV) {
  const int waveId = threadIdx.x >> 5;
  const int lane   = threadIdx.x & 31;
  const int tileM  = blockIdx.x >> 3;                      // uniform per block
  const int tileN  = ((blockIdx.x & 7) << 3) | waveId;
  const int m0 = tileM << 4, n0 = tileN << 4;
  const int half = lane >> 4, l = lane & 15;

  const _Float16* bRow = WT + (size_t)(n0 + l) * D_MODEL + half * 8;
  v8f acc = {};

#if USE_TDM
  __shared__ float abuf[2][16 * 33];                       // padded rows
  const float* aBase = X + (size_t)m0 * D_MODEL;
  const unsigned lds0 = (unsigned)(uintptr_t)&abuf[0][0];
  const unsigned lds1 = (unsigned)(uintptr_t)&abuf[1][0];
  if (threadIdx.x < 32) {
    tdm_load_2d(lds0, aBase, /*log2B*/2, /*cols*/32, /*rows*/16,
                D_MODEL, MROWS, /*padInt:32dw*/4, /*padAmt:1dw*/0, 1);
    wait_tensorcnt0();
  }
  __syncthreads();
  for (int kb = 0; kb < D_MODEL / 32; ++kb) {
    const int cur = kb & 1;
    if (threadIdx.x < 32 && kb + 1 < D_MODEL / 32)
      tdm_load_2d(cur ? lds0 : lds1, aBase + (kb + 1) * 32, 2, 32, 16,
                  D_MODEL, MROWS, 4, 0, 1);
    __builtin_prefetch(bRow + kb * 32 + 64, 0, 1);
    v16h a = pack16_from_f32(&abuf[cur][l * 33 + half * 8]);
    v16h b = pack16_from_f16(bRow + kb * 32);
    acc = wmma_acc(a, b, acc);
    if (threadIdx.x < 32) wait_tensorcnt0();
    __syncthreads();
  }
#else
  const float* aRow = X + (size_t)(m0 + l) * D_MODEL + half * 8;
  for (int kb = 0; kb < D_MODEL / 32; ++kb) {
    const int k = kb * 32;
    __builtin_prefetch(aRow + k + 64, 0, 1);
    v16h a = pack16_from_f32(aRow + k);
    v16h b = pack16_from_f16(bRow + k);
    acc = wmma_acc(a, b, acc);
  }
#endif

  const float bn = bias[n0 + l];
  const int n = n0 + l;
  const int h = n >> 6, dh = n & 63;
#pragma unroll
  for (int r = 0; r < 8; ++r) {
    const int m = m0 + r + 8 * half;          // flattened (b,s) row
    const int bb = m >> 11, s = m & 2047;
    const float vv = acc[r] + bn;
    if (!transposedV)
      dst[((size_t)(bb * NHEAD + h) * SEQ + s) * DHEAD + dh] = (_Float16)vv;
    else
      dst[((size_t)(bb * NHEAD + h) * DHEAD + dh) * SEQ + s] = (_Float16)vv;
  }
}

// ---------------------------------------------------------------------------
// Kernel 2: scores[bh, q, k] = q_proj[bh, q, :] . k_proj[bh, k, :]
// K-dim is only 64 (2 WMMAs) -> direct global b128 loads, no staging.
// ---------------------------------------------------------------------------
__global__ __launch_bounds__(256) void scores_kernel(const _Float16* __restrict__ qP,
                                                     const _Float16* __restrict__ kP,
                                                     float* __restrict__ attn) {
  const int w    = (blockIdx.x * blockDim.x + threadIdx.x) >> 5;
  const int lane = threadIdx.x & 31;
  const int bh = w >> 14;              // / (128*128)
  const int t  = w & 16383;
  const int tq = t >> 7, tk = t & 127;
  const int q0 = tq << 4, k0 = tk << 4;
  const int half = lane >> 4, l = lane & 15;

  const _Float16* aRow = qP + ((size_t)bh * SEQ + q0 + l) * DHEAD + half * 8;
  const _Float16* bRow = kP + ((size_t)bh * SEQ + k0 + l) * DHEAD + half * 8;

  v8f acc = {};
#pragma unroll
  for (int kb = 0; kb < DHEAD / 32; ++kb) {
    const int k = kb * 32;
    v16h a = pack16_from_f16(aRow + k);
    v16h b = pack16_from_f16(bRow + k);
    acc = wmma_acc(a, b, acc);
  }
#pragma unroll
  for (int r = 0; r < 8; ++r) {
    const int qq = q0 + r + 8 * half;
    attn[((size_t)bh * SEQ + qq) * SEQ + k0 + l] = acc[r];
  }
}

// ---------------------------------------------------------------------------
// Kernel 3: in-place masked, scaled softmax over each attn row (2048 wide).
// ---------------------------------------------------------------------------
__global__ __launch_bounds__(256) void softmax_kernel(float* __restrict__ attn,
                                                      const int* __restrict__ mask) {
  __shared__ float red[256];
  const int row = blockIdx.x;               // 0 .. B*H*S-1
  const int bh  = row >> 11;
  const int b   = bh >> 4;
  const int q   = row & 2047;
  float*     p    = attn + (size_t)row * SEQ;
  const int* mrow = mask + ((size_t)b * SEQ + q) * SEQ;   // mask[B,1,S,S]
  const int  t    = threadIdx.x;

  float vals[8];
  float lmax = -3.0e38f;
#pragma unroll
  for (int i = 0; i < 8; ++i) {
    const int c = t + i * 256;
    float s = p[c] * 0.125f;                // 1/sqrt(DHEAD)
    if (mrow[c] == 0) s = -1.0e9f;
    vals[i] = s;
    lmax = fmaxf(lmax, s);
  }
  red[t] = lmax; __syncthreads();
  for (int off = 128; off > 0; off >>= 1) {
    if (t < off) red[t] = fmaxf(red[t], red[t + off]);
    __syncthreads();
  }
  const float rmax = red[0];
  __syncthreads();

  float lsum = 0.0f;
#pragma unroll
  for (int i = 0; i < 8; ++i) { vals[i] = __expf(vals[i] - rmax); lsum += vals[i]; }
  red[t] = lsum; __syncthreads();
  for (int off = 128; off > 0; off >>= 1) {
    if (t < off) red[t] += red[t + off];
    __syncthreads();
  }
  const float inv = 1.0f / red[0];
#pragma unroll
  for (int i = 0; i < 8; ++i) p[t + i * 256] = vals[i] * inv;
}

// ---------------------------------------------------------------------------
// Kernel 4: ctx = attn[bh, S, S] @ v ; vT f16 [B,H,DHEAD,S].
// Block = 8 waves covering 2 tm x 4 tn tiles -> TDM-stage 32 attn rows
// (32x32 fp32 per K step, padded stride 33) with double buffering; this is
// the dominant HBM stream (attn re-read), so staging cuts it 4x.
// ---------------------------------------------------------------------------
__global__ __launch_bounds__(256) void context_kernel(const float* __restrict__ attn,
                                                      const _Float16* __restrict__ vT,
                                                      _Float16* __restrict__ ctx) {
  const int waveId = threadIdx.x >> 5;
  const int lane   = threadIdx.x & 31;
  const int bh = blockIdx.x >> 6;                // 64 blocks per (b,h)
  const int bi = blockIdx.x & 63;                // uniform per block
  const int tmLocal = waveId >> 2;               // 0..1
  const int tn      = waveId & 3;                // 0..3
  const int m0 = bi * 32 + tmLocal * 16;
  const int n0 = tn << 4;
  const int half = lane >> 4, l = lane & 15;

  const _Float16* bRow = vT + ((size_t)bh * DHEAD + n0 + l) * SEQ + half * 8;
  v8f acc = {};

#if USE_TDM
  __shared__ float cbuf[2][32 * 33];
  const float* aTile = attn + ((size_t)bh * SEQ + bi * 32) * SEQ;  // 32-row stripe
  const unsigned lds0 = (unsigned)(uintptr_t)&cbuf[0][0];
  const unsigned lds1 = (unsigned)(uintptr_t)&cbuf[1][0];
  if (threadIdx.x < 32) {
    tdm_load_2d(lds0, aTile, 2, 32, 32, SEQ, SEQ, 4, 0, 1);
    wait_tensorcnt0();
  }
  __syncthreads();
  const int rloc = tmLocal * 16 + l;
  for (int kb = 0; kb < SEQ / 32; ++kb) {
    const int cur = kb & 1;
    if (threadIdx.x < 32 && kb + 1 < SEQ / 32)
      tdm_load_2d(cur ? lds0 : lds1, aTile + (kb + 1) * 32, 2, 32, 32,
                  SEQ, SEQ, 4, 0, 1);
    __builtin_prefetch(bRow + kb * 32 + 64, 0, 1);
    v16h a = pack16_from_f32(&cbuf[cur][rloc * 33 + half * 8]);
    v16h b = pack16_from_f16(bRow + kb * 32);
    acc = wmma_acc(a, b, acc);
    if (threadIdx.x < 32) wait_tensorcnt0();
    __syncthreads();
  }
#else
  const float* aRow = attn + ((size_t)bh * SEQ + m0 + l) * SEQ + half * 8;
  for (int kb = 0; kb < SEQ / 32; ++kb) {
    const int k = kb * 32;
    __builtin_prefetch(aRow + k + 64, 0, 1);
    v16h a = pack16_from_f32(aRow + k);
    v16h b = pack16_from_f16(bRow + k);
    acc = wmma_acc(a, b, acc);
  }
#endif

  const int bb = bh >> 4, h = bh & 15;
  const int col = h * DHEAD + n0 + l;
#pragma unroll
  for (int r = 0; r < 8; ++r) {
    const int s = m0 + r + 8 * half;
    ctx[((size_t)(bb * SEQ + s)) * D_MODEL + col] = (_Float16)acc[r];
  }
}

// ---------------------------------------------------------------------------
// Kernel 5: out = ctx[4096,1024] @ Wo + bo  (fp32 result into d_out)
// TDM-stages the f16 ctx A-tiles (16x32 halfs, pad 1 dword => stride 34).
// ---------------------------------------------------------------------------
__global__ __launch_bounds__(256) void outproj_kernel(const _Float16* __restrict__ ctx,
                                                      const _Float16* __restrict__ WoT,
                                                      const float* __restrict__ bo,
                                                      float* __restrict__ out) {
  const int waveId = threadIdx.x >> 5;
  const int lane   = threadIdx.x & 31;
  const int tileM  = blockIdx.x >> 3;                      // uniform per block
  const int tileN  = ((blockIdx.x & 7) << 3) | waveId;
  const int m0 = tileM << 4, n0 = tileN << 4;
  const int half = lane >> 4, l = lane & 15;

  const _Float16* bRow = WoT + (size_t)(n0 + l) * D_MODEL + half * 8;
  v8f acc = {};

#if USE_TDM
  __shared__ _Float16 hbuf[2][16 * 34];                    // 32 halfs + 1 dword pad
  const _Float16* aBase = ctx + (size_t)m0 * D_MODEL;
  const unsigned lds0 = (unsigned)(uintptr_t)&hbuf[0][0];
  const unsigned lds1 = (unsigned)(uintptr_t)&hbuf[1][0];
  if (threadIdx.x < 32) {
    tdm_load_2d(lds0, aBase, /*log2B*/1, 32, 16, D_MODEL, MROWS,
                /*padInt:16dw*/3, /*padAmt:1dw*/0, 1);
    wait_tensorcnt0();
  }
  __syncthreads();
  for (int kb = 0; kb < D_MODEL / 32; ++kb) {
    const int cur = kb & 1;
    if (threadIdx.x < 32 && kb + 1 < D_MODEL / 32)
      tdm_load_2d(cur ? lds0 : lds1, aBase + (kb + 1) * 32, 1, 32, 16,
                  D_MODEL, MROWS, 3, 0, 1);
    __builtin_prefetch(bRow + kb * 32 + 64, 0, 1);
    v16h a = pack16_from_f16(&hbuf[cur][l * 34 + half * 8]);
    v16h b = pack16_from_f16(bRow + kb * 32);
    acc = wmma_acc(a, b, acc);
    if (threadIdx.x < 32) wait_tensorcnt0();
    __syncthreads();
  }
#else
  const _Float16* aRow = ctx + (size_t)(m0 + l) * D_MODEL + half * 8;
  for (int kb = 0; kb < D_MODEL / 32; ++kb) {
    const int k = kb * 32;
    __builtin_prefetch(aRow + k + 64, 0, 1);
    v16h a = pack16_from_f16(aRow + k);
    v16h b = pack16_from_f16(bRow + k);
    acc = wmma_acc(a, b, acc);
  }
#endif

  const float bn = bo[n0 + l];
#pragma unroll
  for (int r = 0; r < 8; ++r) {
    const int m = m0 + r + 8 * half;
    out[(size_t)m * D_MODEL + n0 + l] = acc[r] + bn;
  }
}

// ---------------------------------------------------------------------------
extern "C" void kernel_launch(void* const* d_in, const int* in_sizes, int n_in,
                              void* d_out, int out_size, void* d_ws, size_t ws_size,
                              hipStream_t stream) {
  (void)in_sizes; (void)n_in; (void)out_size; (void)ws_size;

  const float* Q  = (const float*)d_in[0];
  const float* K  = (const float*)d_in[1];
  const float* V  = (const float*)d_in[2];
  const float* Wq = (const float*)d_in[3];
  const float* bq = (const float*)d_in[4];
  const float* Wk = (const float*)d_in[5];
  const float* bk = (const float*)d_in[6];
  const float* Wv = (const float*)d_in[7];
  const float* bv = (const float*)d_in[8];
  const float* Wo = (const float*)d_in[9];
  const float* bo = (const float*)d_in[10];
  const int* mask = (const int*)d_in[11];

  float* out  = (float*)d_out;                          // [B,S,D] fp32
  float* attn = out + (size_t)BATCH * SEQ * D_MODEL;    // [B,H,S,S] fp32

  // Workspace layout (f16): 4x WT (2 MiB each) + qP/kP/vT/ctx (8 MiB each) = 40 MiB
  const size_t WSZ = (size_t)D_MODEL * D_MODEL;         // elements
  const size_t PSZ = (size_t)BATCH * SEQ * D_MODEL;     // elements
  _Float16* WqT = (_Float16*)d_ws;
  _Float16* WkT = WqT + WSZ;
  _Float16* WvT = WkT + WSZ;
  _Float16* WoT = WvT + WSZ;
  _Float16* qP  = WoT + WSZ;   // [B,H,S,DHEAD]
  _Float16* kP  = qP + PSZ;    // [B,H,S,DHEAD]
  _Float16* vT  = kP + PSZ;    // [B,H,DHEAD,S]
  _Float16* ctx = vT + PSZ;    // [B,S,D]

  wt_f16_kernel<<<1024, 256, 0, stream>>>(Wq, WqT);
  wt_f16_kernel<<<1024, 256, 0, stream>>>(Wk, WkT);
  wt_f16_kernel<<<1024, 256, 0, stream>>>(Wv, WvT);
  wt_f16_kernel<<<1024, 256, 0, stream>>>(Wo, WoT);

  proj_kernel<<<2048, 256, 0, stream>>>(Q, WqT, bq, qP, 0);
  proj_kernel<<<2048, 256, 0, stream>>>(K, WkT, bk, kP, 0);
  proj_kernel<<<2048, 256, 0, stream>>>(V, WvT, bv, vT, 1);

  scores_kernel<<<65536, 256, 0, stream>>>(qP, kP, attn);
  softmax_kernel<<<65536, 256, 0, stream>>>(attn, mask);
  context_kernel<<<2048, 256, 0, stream>>>(attn, vT, ctx);
  outproj_kernel<<<2048, 256, 0, stream>>>(ctx, WoT, bo, out);
}
